// BaichuanAttention_45208825757929
// MI455X (gfx1250) — matrix operand order repaired and checked by reading, not compiled, not run on hardware
//
#include <hip/hip_runtime.h>

// ---------------------------------------------------------------------------
// Baichuan attention forward on gfx1250 (MI455X), bf16 WMMA + f32 accumulate.
// B=2, S=2048, H=4096, NH=32, HD=128.
// Flash-attention V tiles are staged into LDS via the Tensor Data Mover
// (tensor_load_to_lds + s_wait_tensorcnt), overlapped with score WMMAs.
// ---------------------------------------------------------------------------

typedef __attribute__((ext_vector_type(16))) __bf16 v16bf;
typedef __attribute__((ext_vector_type(8)))  float  v8f;
typedef __attribute__((ext_vector_type(4)))  unsigned int u32x4;
typedef __attribute__((ext_vector_type(8)))  unsigned int u32x8;

#define B_   2
#define S_   2048
#define H_   4096
#define NH_  32
#define HD_  128
#define BS_  (B_ * S_)      // 4096 rows total
#define H3_  (3 * H_)       // 12288

static __device__ inline v8f vzero8() {
  v8f z = {0.f, 0.f, 0.f, 0.f, 0.f, 0.f, 0.f, 0.f};
  return z;
}

static __device__ inline v8f wmma_bf16(v16bf a, v16bf b, v8f c) {
  // D = A(16x32 bf16) x B(32x16 bf16) + C(16x16 f32)
  return __builtin_amdgcn_wmma_f32_16x16x32_bf16(
      /*neg_a=*/false, a, /*neg_b=*/false, b,
      /*c_mod=*/(short)0, c, /*reuse_a=*/false, /*reuse_b=*/false);
}

// A-fragment (16x32, bf16, row-major source with full K=32 row at `row`).
// Lanes hold M=lane&15, VGPR j -> K=16*half+2j,2j+1; half = lane>>4.
static __device__ inline v16bf frag_a_rowmajor(const __bf16* __restrict__ row,
                                               int half) {
  v16bf a;
#pragma unroll
  for (int j = 0; j < 8; ++j) {
    const int kb = (half << 4) + (j << 1);
    a[2 * j]     = row[kb];
    a[2 * j + 1] = row[kb + 1];
  }
  return a;
}

// B-fragment (32x16) where B[k][n] = W[n0+n][k0+k], W row-major (K contiguous).
// col = &W[(n0 + (lane&15)) * ldw + k0]. Lanes 0-15: K=0..15, lanes 16-31: K=16..31.
static __device__ inline v16bf frag_b_from_wT(const __bf16* __restrict__ col,
                                              int half) {
  v16bf b;
#pragma unroll
  for (int j = 0; j < 8; ++j) {
    const int kb = (half << 4) + (j << 1);
    b[2 * j]     = col[kb];
    b[2 * j + 1] = col[kb + 1];
  }
  return b;
}

// B-fragment (32x16) where B[k][n] = M[k0+k][n0+n], M row-major, leading dim ld.
// base = &M[k0*ld + n0 + (lane&15)].
static __device__ inline v16bf frag_b_rowmajor(const __bf16* __restrict__ base,
                                               int ld, int half) {
  v16bf b;
#pragma unroll
  for (int j = 0; j < 8; ++j) {
    const int kk = (half << 4) + (j << 1);
    b[2 * j]     = base[kk * ld];
    b[2 * j + 1] = base[(kk + 1) * ld];
  }
  return b;
}

// ---------------------------------------------------------------------------
// Tensor Data Mover: DMA a [32 x HD_] bf16 tile (row-major, stride HD_) from
// global memory into LDS at byte offset lds_off. D# per CDNA5 ISA:
//   group0: count=1, lds_addr, global_addr(57b), type=2
//   group1: data_size=2B, tensor_dim0=128, tensor_dim1=32, tile 128x32,
//           tensor_dim0_stride=128
// 2-group form (VADDR2/VADDR3 = NULL) => tensors up to 2D. Completion is
// tracked with TENSORcnt.
// ---------------------------------------------------------------------------
static __device__ inline void tdm_load_tile_32xHD(const __bf16* gsrc,
                                                  unsigned lds_off) {
  const unsigned long long ga = (unsigned long long)(uintptr_t)gsrc;
  u32x4 g0;
  g0[0] = 1u;                                   // count=1 (valid), user mode
  g0[1] = lds_off;                              // lds_addr (bytes)
  g0[2] = (unsigned)(ga & 0xffffffffu);         // global_addr[31:0]
  g0[3] = (unsigned)((ga >> 32) & 0x01ffffffu)  // global_addr[56:32]
          | (2u << 30);                         // type=2 ("image")
  u32x8 g1;
  g1[0] = 0x00010000u;                          // data_size=1 -> 2 bytes
  g1[1] = ((unsigned)HD_ & 0xffffu) << 16;      // tensor_dim0[15:0]=128
  g1[2] = (32u & 0xffffu) << 16;                // tensor_dim1[15:0]=32
  g1[3] = ((unsigned)HD_ & 0xffffu) << 16;      // tile_dim0=128
  g1[4] = 32u;                                  // tile_dim1=32
  g1[5] = (unsigned)HD_;                        // tensor_dim0_stride=128
  g1[6] = 0u;
  g1[7] = 0u;
  asm volatile("tensor_load_to_lds %0, %1" : : "s"(g0), "s"(g1) : "memory");
}

// ---------------------------------------------------------------------------
// Kernel 0: f32 -> bf16 conversion
// ---------------------------------------------------------------------------
__global__ void cvt_f32_bf16_kernel(const float* __restrict__ in,
                                    __bf16* __restrict__ out, int n) {
  int i = blockIdx.x * blockDim.x + threadIdx.x;
  if (i < n) out[i] = (__bf16)in[i];
}

// ---------------------------------------------------------------------------
// Kernel 1: fused QKV projection + RoPE.
// One wave computes a 16(M) x 128(N) strip of qkv = x @ w_pack^T.
// The 128-col strip is exactly one head => rotate_half partner (hd ^ 64) is
// chunk c^4 in the same lane; RoPE is lane-local.
// ---------------------------------------------------------------------------
__global__ __launch_bounds__(256) void qkv_rope_kernel(
    const __bf16* __restrict__ xb,   // [BS_, H_]
    const __bf16* __restrict__ wpb,  // [H3_, H_]
    __bf16* __restrict__ Qb,         // [B, NH, S, HD] (RoPE applied)
    __bf16* __restrict__ Kb,         // [B, NH, S, HD] (RoPE applied)
    __bf16* __restrict__ Vb) {       // [B, NH, S, HD]
  const int lane = threadIdx.x & 31;
  const int wid  = blockIdx.x * (blockDim.x >> 5) + (threadIdx.x >> 5);
  const int NT   = H3_ / 128;  // 96 column strips
  const int mt   = wid / NT;
  const int nt   = wid % NT;
  const int n    = lane & 15;
  const int half = lane >> 4;

  v8f acc[8];
#pragma unroll
  for (int c = 0; c < 8; ++c) acc[c] = vzero8();

  const __bf16* arow     = xb + (size_t)(mt * 16 + n) * H_;
  const __bf16* wcolbase = wpb + (size_t)(nt * 128) * H_;

  for (int k0 = 0; k0 < H_; k0 += 32) {
    __builtin_prefetch(arow + k0 + 128, 0, 1);  // global_prefetch_b8
    v16bf a = frag_a_rowmajor(arow + k0, half);
#pragma unroll
    for (int c = 0; c < 8; ++c) {
      const __bf16* col = wcolbase + (size_t)(c * 16 + n) * H_ + k0;
      v16bf b = frag_b_from_wT(col, half);
      acc[c] = wmma_bf16(a, b, acc[c]);
    }
  }

  // Epilogue: split into Q/K/V segment, apply RoPE to Q,K, scatter bf16.
  const int e0   = nt * 128;
  const int seg  = e0 / H_;            // 0=Q 1=K 2=V
  const int head = (e0 % H_) / HD_;
  __bf16* dst = (seg == 0) ? Qb : ((seg == 1) ? Kb : Vb);
  const bool doRope = (seg < 2);
  const float kLog1e4Over64 = 0.14391156f;  // ln(10000)/64

#pragma unroll
  for (int r = 0; r < 8; ++r) {
    const int m    = r + 8 * half;
    const int Mrow = mt * 16 + m;
    const int bb   = Mrow / S_;
    const int s    = Mrow % S_;
    __bf16* drow = dst + ((size_t)(bb * NH_ + head) * S_ + s) * HD_;
#pragma unroll
    for (int c = 0; c < 8; ++c) {
      const int hd = c * 16 + n;
      float v = acc[c][r];
      if (doRope) {
        const float part = acc[c ^ 4][r];
        const float rot  = (c < 4) ? -part : part;
        const float freq = __expf(-(float)(hd & 63) * kLog1e4Over64);
        float sn, cs;
        __sincosf((float)s * freq, &sn, &cs);
        v = v * cs + rot * sn;
      }
      drow[hd] = (__bf16)v;
    }
  }
}

// ---------------------------------------------------------------------------
// Kernel 2: causal flash attention. One wave per (b, head, 16-row q tile).
// Streams 32-row K/V tiles. V tile is DMA'd into LDS by the TDM while the
// score WMMAs + online softmax run; P@V reads V fragments from LDS after
// s_wait_tensorcnt 0.
// ---------------------------------------------------------------------------
__global__ __launch_bounds__(32) void flash_attn_kernel(
    const __bf16* __restrict__ Qb, const __bf16* __restrict__ Kb,
    const __bf16* __restrict__ Vb, __bf16* __restrict__ ctx) {  // ctx: [BS_, H_]
  __shared__ __bf16 Pl[16 * 32];     // P tile,  row-major 16x32
  __shared__ __bf16 Vl[32 * HD_];    // V tile,  row-major 32x128 (TDM dest)

  const int lane = threadIdx.x;
  const int qt   = blockIdx.x % (S_ / 16);
  const int bh   = blockIdx.x / (S_ / 16);  // b*NH + head
  const int n    = lane & 15;
  const int half = lane >> 4;
  const int q0   = qt * 16;

  const __bf16* Qh = Qb + (size_t)bh * S_ * HD_;
  const __bf16* Kh = Kb + (size_t)bh * S_ * HD_;
  const __bf16* Vh = Vb + (size_t)bh * S_ * HD_;

  const unsigned vl_off = (unsigned)(uintptr_t)(void*)&Vl[0];  // LDS byte offset

  // Hoist Q A-fragments (reused across all K tiles): HD=128 -> 4 K-chunks.
  v16bf qf[4];
#pragma unroll
  for (int c = 0; c < 4; ++c)
    qf[c] = frag_a_rowmajor(Qh + (size_t)(q0 + n) * HD_ + c * 32, half);

  v8f acc[8];  // ctx accumulator: 16 x 128 = 8 C-tiles
#pragma unroll
  for (int c = 0; c < 8; ++c) acc[c] = vzero8();
  float mrun[8], lrun[8];
#pragma unroll
  for (int r = 0; r < 8; ++r) { mrun[r] = -3.0e38f; lrun[r] = 0.f; }

  const float scale = 0.08838834764831845f;  // 1/sqrt(HD)

  for (int kt0 = 0; kt0 <= q0 + 15; kt0 += 32) {
    // Prior iteration's LDS reads must finish before the TDM overwrites Vl.
    __syncthreads();
    // Kick off the async V-tile DMA; it runs under the score computation.
    tdm_load_tile_32xHD(Vh + (size_t)kt0 * HD_, vl_off);

    // scores tile 16x32 = two 16x16 C tiles, K=HD chained over 4 chunks
    v8f s0 = vzero8(), s1 = vzero8();
#pragma unroll
    for (int c = 0; c < 4; ++c) {
      v16bf kb0 = frag_b_from_wT(Kh + (size_t)(kt0 + n) * HD_ + c * 32, half);
      s0 = wmma_bf16(qf[c], kb0, s0);
      v16bf kb1 = frag_b_from_wT(Kh + (size_t)(kt0 + 16 + n) * HD_ + c * 32, half);
      s1 = wmma_bf16(qf[c], kb1, s1);
    }

    // scale + causal mask (C layout: lane -> col n, vgpr r -> row r+8*half)
#pragma unroll
    for (int r = 0; r < 8; ++r) {
      const int row = q0 + r + 8 * half;
      float v0 = s0[r] * scale;
      float v1 = s1[r] * scale;
      if (kt0 + n > row)      v0 = -3.0e38f;
      if (kt0 + 16 + n > row) v1 = -3.0e38f;
      s0[r] = v0; s1[r] = v1;
    }

    // online softmax: per-row reduction across the 16 lanes of this half
    float alpha[8];
#pragma unroll
    for (int r = 0; r < 8; ++r) {
      float mx = fmaxf(s0[r], s1[r]);
#pragma unroll
      for (int off = 1; off < 16; off <<= 1) mx = fmaxf(mx, __shfl_xor(mx, off, 32));
      const float mnew = fmaxf(mrun[r], mx);
      const float p0 = __expf(s0[r] - mnew);
      const float p1 = __expf(s1[r] - mnew);
      float rs = p0 + p1;
#pragma unroll
      for (int off = 1; off < 16; off <<= 1) rs += __shfl_xor(rs, off, 32);
      alpha[r] = __expf(mrun[r] - mnew);
      lrun[r]  = lrun[r] * alpha[r] + rs;
      mrun[r]  = mnew;
      s0[r] = p0; s1[r] = p1;
    }
#pragma unroll
    for (int c = 0; c < 8; ++c)
#pragma unroll
      for (int r = 0; r < 8; ++r) acc[c][r] *= alpha[r];

    // P (C layout, f32) -> LDS row-major bf16 -> A-fragment layout
#pragma unroll
    for (int r = 0; r < 8; ++r) {
      const int m = r + 8 * half;
      Pl[m * 32 + n]      = (__bf16)s0[r];
      Pl[m * 32 + 16 + n] = (__bf16)s1[r];
    }
    __syncthreads();
    v16bf pf = frag_a_rowmajor(Pl + n * 32, half);

    // V tile must be resident in LDS before the P@V fragments are read.
    __builtin_amdgcn_s_wait_tensorcnt(0);

    // ctx += P(16x32) @ V(32x128), V fragments served from LDS
#pragma unroll
    for (int c = 0; c < 8; ++c) {
      v16bf vb = frag_b_rowmajor(&Vl[0] + c * 16 + n, HD_, half);
      acc[c] = wmma_bf16(pf, vb, acc[c]);
    }
  }

  // normalize + store ctx as bf16 [B*S, H] for the output projection
  const int bb   = bh / NH_;
  const int head = bh % NH_;
#pragma unroll
  for (int r = 0; r < 8; ++r) {
    const int m = r + 8 * half;
    const int s = q0 + m;
    const float inv_l = 1.f / lrun[r];
    __bf16* drow = ctx + (size_t)(bb * S_ + s) * H_ + head * HD_;
#pragma unroll
    for (int c = 0; c < 8; ++c) drow[c * 16 + n] = (__bf16)(acc[c][r] * inv_l);
  }
}

// ---------------------------------------------------------------------------
// Kernel 3: output projection out = ctx @ w_o^T, f32 output.
// One wave computes a 16(M) x 64(N) tile.
// ---------------------------------------------------------------------------
__global__ __launch_bounds__(256) void out_proj_kernel(
    const __bf16* __restrict__ ctx,  // [BS_, H_]
    const __bf16* __restrict__ wob,  // [H_, H_]
    float* __restrict__ out) {       // [BS_, H_]
  const int lane = threadIdx.x & 31;
  const int wid  = blockIdx.x * (blockDim.x >> 5) + (threadIdx.x >> 5);
  const int NT   = H_ / 64;  // 64
  const int mt   = wid / NT;
  const int nt   = wid % NT;
  const int n    = lane & 15;
  const int half = lane >> 4;

  v8f acc[4];
#pragma unroll
  for (int c = 0; c < 4; ++c) acc[c] = vzero8();

  const __bf16* arow = ctx + (size_t)(mt * 16 + n) * H_;
  const __bf16* wb   = wob + (size_t)(nt * 64) * H_;

  for (int k0 = 0; k0 < H_; k0 += 32) {
    __builtin_prefetch(arow + k0 + 128, 0, 1);
    v16bf a = frag_a_rowmajor(arow + k0, half);
#pragma unroll
    for (int c = 0; c < 4; ++c) {
      v16bf b = frag_b_from_wT(wb + (size_t)(c * 16 + n) * H_ + k0, half);
      acc[c] = wmma_bf16(a, b, acc[c]);
    }
  }

#pragma unroll
  for (int c = 0; c < 4; ++c)
#pragma unroll
    for (int r = 0; r < 8; ++r) {
      const int m = r + 8 * half;
      out[(size_t)(mt * 16 + m) * H_ + nt * 64 + c * 16 + n] = acc[c][r];
    }
}

// ---------------------------------------------------------------------------
// Launch
// ---------------------------------------------------------------------------
extern "C" void kernel_launch(void* const* d_in, const int* in_sizes, int n_in,
                              void* d_out, int out_size, void* d_ws,
                              size_t ws_size, hipStream_t stream) {
  (void)in_sizes; (void)n_in; (void)out_size; (void)ws_size;

  const float* x_f   = (const float*)d_in[0];  // [B,S,H]
  const float* wp_f  = (const float*)d_in[1];  // [3H,H]
  const float* wo_f  = (const float*)d_in[2];  // [H,H]
  float*       out_f = (float*)d_out;          // [B,S,H]

  char* w = (char*)d_ws;
  const size_t xb_bytes  = (size_t)BS_ * H_ * 2;
  const size_t wpb_bytes = (size_t)H3_ * H_ * 2;
  const size_t wob_bytes = (size_t)H_ * H_ * 2;
  const size_t qkv_bytes = (size_t)B_ * NH_ * S_ * HD_ * 2;
  const size_t ctx_bytes = (size_t)BS_ * H_ * 2;

  __bf16* xb  = (__bf16*)w;  w += xb_bytes;
  __bf16* wpb = (__bf16*)w;  w += wpb_bytes;
  __bf16* wob = (__bf16*)w;  w += wob_bytes;
  __bf16* Qb  = (__bf16*)w;  w += qkv_bytes;
  __bf16* Kb  = (__bf16*)w;  w += qkv_bytes;
  __bf16* Vb  = (__bf16*)w;  w += qkv_bytes;
  __bf16* ctx = (__bf16*)w;  w += ctx_bytes;

  // 0) convert inputs to bf16
  {
    const int nx = BS_ * H_;
    cvt_f32_bf16_kernel<<<(nx + 255) / 256, 256, 0, stream>>>(x_f, xb, nx);
    const int nwp = H3_ * H_;
    cvt_f32_bf16_kernel<<<(nwp + 255) / 256, 256, 0, stream>>>(wp_f, wpb, nwp);
    const int nwo = H_ * H_;
    cvt_f32_bf16_kernel<<<(nwo + 255) / 256, 256, 0, stream>>>(wo_f, wob, nwo);
  }

  // 1) fused QKV + RoPE: (BS_/16) * (H3_/128) waves, 8 waves / block
  {
    const int waves  = (BS_ / 16) * (H3_ / 128);  // 24576
    const int blocks = waves / 8;                 // 3072
    qkv_rope_kernel<<<blocks, 256, 0, stream>>>(xb, wpb, Qb, Kb, Vb);
  }

  // 2) flash attention: one wave per (b*NH, q-tile)
  {
    const int blocks = B_ * NH_ * (S_ / 16);  // 8192
    flash_attn_kernel<<<blocks, 32, 0, stream>>>(Qb, Kb, Vb, ctx);
  }

  // 3) output projection
  {
    const int waves  = (BS_ / 16) * (H_ / 64);  // 16384
    const int blocks = waves / 8;               // 2048
    out_proj_kernel<<<blocks, 256, 0, stream>>>(ctx, wob, out_f);
  }
}
